// RLSP_dudy_76020921139335
// MI455X (gfx1250) — matrix operand order repaired and checked by reading, not compiled, not run on hardware
//
#include <hip/hip_runtime.h>
#include <hip/hip_bf16.h>

typedef __attribute__((ext_vector_type(16))) _Float16 v16h;
typedef __attribute__((ext_vector_type(8)))  _Float16 v8h;
typedef __attribute__((ext_vector_type(8)))  float    v8f;

#define HGT 180
#define WID 320
#define HP  182
#define WPD 322
#define NPIX (HP*WPD)
#define T_FRAMES 5
#define FH 720
#define FW 1280

union V16U { v16h v; v8h h[2]; };

// ---------------- zero fill ----------------
__global__ void k_zero(float4* __restrict__ p, long n) {
  long i = (long)blockIdx.x * blockDim.x + threadIdx.x;
  long stride = (long)gridDim.x * blockDim.x;
  float4 z = make_float4(0.f, 0.f, 0.f, 0.f);
  for (; i < n; i += stride) p[i] = z;
}

// ---------------- weight / bias packing ----------------
// conv1_w [128][185][3][3] f32 -> [128][9][192] f16 (cin zero-padded)
__global__ void k_pack_w1(const float* __restrict__ w, _Float16* __restrict__ dst) {
  const int n = 128 * 9 * 192;
  for (int i = blockIdx.x * blockDim.x + threadIdx.x; i < n; i += gridDim.x * blockDim.x) {
    int co = i / (9 * 192);
    int rem = i % (9 * 192);
    int tap = rem / 192, ci = rem % 192;
    float v = 0.f;
    if (ci < 185) v = w[((co * 185 + ci) * 3 + tap / 3) * 3 + tap % 3];
    dst[i] = (_Float16)v;
  }
}
// mid_w [5][128][128][3][3] -> [5][128][9][128]
__global__ void k_pack_wm(const float* __restrict__ w, _Float16* __restrict__ dst) {
  const int n = 5 * 128 * 9 * 128;
  for (int i = blockIdx.x * blockDim.x + threadIdx.x; i < n; i += gridDim.x * blockDim.x) {
    int l = i / (128 * 9 * 128);
    int rem = i % (128 * 9 * 128);
    int co = rem / (9 * 128);
    int rem2 = rem % (9 * 128);
    int tap = rem2 / 128, ci = rem2 % 128;
    float v = w[(((l * 128 + co) * 128 + ci) * 3 + tap / 3) * 3 + tap % 3];
    dst[i] = (_Float16)v;
  }
}
// out_w [176][128][3][3] -> [192][9][128] (cout zero-padded)
__global__ void k_pack_wo(const float* __restrict__ w, _Float16* __restrict__ dst) {
  const int n = 192 * 9 * 128;
  for (int i = blockIdx.x * blockDim.x + threadIdx.x; i < n; i += gridDim.x * blockDim.x) {
    int co = i / (9 * 128);
    int rem = i % (9 * 128);
    int tap = rem / 128, ci = rem % 128;
    float v = (co < 176) ? w[((co * 128 + ci) * 3 + tap / 3) * 3 + tap % 3] : 0.f;
    dst[i] = (_Float16)v;
  }
}
// biases: [0,128) conv1, [128,768) mid (5x128), [768,960) out (padded to 192)
__global__ void k_pack_bias(const float* __restrict__ b1, const float* __restrict__ bm,
                            const float* __restrict__ bo, float* __restrict__ dst) {
  int i = blockIdx.x * blockDim.x + threadIdx.x;
  if (i >= 960) return;
  if (i < 128) dst[i] = b1[i];
  else if (i < 768) dst[i] = bm[i - 128];
  else { int j = i - 768; dst[i] = (j < 176) ? bo[j] : 0.f; }
}

// ---------------- frame packing: channels 0..8 of actA ----------------
__global__ void k_pack_frames(const float* __restrict__ x, _Float16* __restrict__ actA, int t) {
  int idx = blockIdx.x * blockDim.x + threadIdx.x;
  const int n = 9 * HGT * WID;
  if (idx >= n) return;
  int c = idx / (HGT * WID);
  int rem = idx % (HGT * WID);
  int h = rem / WID, w = rem % WID;
  int fsel = c / 3, ch = c % 3;
  int fi = t + fsel - 1;
  fi = fi < 0 ? 0 : (fi > T_FRAMES - 1 ? T_FRAMES - 1 : fi);
  float v = x[(fi * 3 + ch) * (HGT * WID) + rem];
  actA[((long)(h + 1) * WPD + (w + 1)) * 192 + c] = (_Float16)v;
}

// ---------------- gaussian feedback path ----------------
__device__ __forceinline__ void gauss13(float* kw) {
  float s = 0.f;
#pragma unroll
  for (int i = 0; i < 13; i++) { float tt = (i - 6) / 1.5f; kw[i] = __expf(-0.5f * tt * tt); s += kw[i]; }
  float inv = 1.f / s;
#pragma unroll
  for (int i = 0; i < 13; i++) kw[i] *= inv;
}

__global__ void k_blur_h(const float* __restrict__ src, float* __restrict__ dst) {
  int idx = blockIdx.x * blockDim.x + threadIdx.x;
  const int n = 3 * FH * FW;
  if (idx >= n) return;
  int c = idx / (FH * FW);
  int rem = idx % (FH * FW);
  int y = rem / FW, xq = rem % FW;
  float kw[13]; gauss13(kw);
  float s = 0.f;
#pragma unroll
  for (int i = 0; i < 13; i++) {
    int xx = xq + i - 6;
    if (xx < 0) xx = -1 - xx;
    if (xx >= FW) xx = 2 * FW - 1 - xx;
    s += kw[i] * src[(c * FH + y) * FW + xx];
  }
  dst[idx] = s;
}

// vertical blur + clip + rint + contiguous-reshape down into actA channels 137..184
__global__ void k_blur_vdown(const float* __restrict__ src, _Float16* __restrict__ actA) {
  int idx = blockIdx.x * blockDim.x + threadIdx.x;
  const int n = 3 * FH * FW;
  if (idx >= n) return;
  int c = idx / (FH * FW);
  int rem = idx % (FH * FW);
  int y = rem / FW, xq = rem % FW;
  float kw[13]; gauss13(kw);
  float s = 0.f;
#pragma unroll
  for (int i = 0; i < 13; i++) {
    int yy = y + i - 6;
    if (yy < 0) yy = -1 - yy;
    if (yy >= FH) yy = 2 * FH - 1 - yy;
    s += kw[i] * src[(c * FH + yy) * FW + xq];
  }
  float r = rintf(fminf(fmaxf(s, 0.f), 255.f));
  int c2off = y / 45;                 // (y*1280+x)/57600
  int hh = (y % 45) * 4 + xq / WID;
  int ww = xq % WID;
  actA[((long)(hh + 1) * WPD + (ww + 1)) * 192 + 137 + c * 16 + c2off] = (_Float16)r;
}

// ---------------- res_add + shuffle_up ----------------
__global__ void k_out_process(const _Float16* __restrict__ y, float* __restrict__ dst) {
  int idx = blockIdx.x * blockDim.x + threadIdx.x;
  if (idx >= HGT * WID) return;
  int h = idx / WID, w = idx % WID;
  const _Float16* p = y + ((long)(h + 1) * WPD + (w + 1)) * 192;
  float r = (float)p[1] * (219.859f / 256.0f) + 16.0f;   // sum(RGB2Y)/256
#pragma unroll
  for (int c2 = 0; c2 < 48; c2++) {
    float v = (float)p[c2] + r;
    int cch = c2 >> 4, k = c2 & 15;
    int Y = k * 45 + (h >> 2);
    int X = (h & 3) * WID + w;
    dst[((long)cch * FH + Y) * FW + X] = v;
  }
}

// ---------------- implicit-GEMM 3x3 conv via WMMA f16 ----------------
// in:  f16 channels-last, padded halo [HP][WPD][ics], valid Cin channels
// wq:  f16 [coutPad][9][Cin]
// out: f16 channels-last padded [HP][WPD][ocs], written at channel ocoff+..
// grid = (WID/64, HGT, coutPad/64), block = 256 (8 waves)
__global__ __launch_bounds__(256)
void k_conv3x3_wmma(const _Float16* __restrict__ in, int ics, int Cin,
                    const _Float16* __restrict__ wq, const float* __restrict__ bias,
                    _Float16* __restrict__ outp, int ocs, int ocoff, int relu) {
  const int lane = threadIdx.x & 31;
  const int wid  = threadIdx.x >> 5;
  const int wx   = blockIdx.x * 64 + (wid & 3) * 16;   // pixel tile (16 px along W)
  const int cb   = blockIdx.z * 64 + (wid >> 2) * 32;  // wave covers cout [cb, cb+32)
  const int h    = blockIdx.y;
  const int ml   = lane & 15;
  const int hi   = lane >> 4;                          // 0 / 1 lane half

  v8f acc0 = {};
  v8f acc1 = {};
#pragma unroll
  for (int tap = 0; tap < 9; ++tap) {
    const int dy = tap / 3 - 1, dx = tap % 3 - 1;
    // A fragment base: 16 pixels (row-major), K-slice of channels
    const _Float16* arow = in + ((long)(h + 1 + dy) * WPD + (wx + ml + 1 + dx)) * ics + hi * 16;
    // B fragment bases for the two 16-wide cout tiles
    const _Float16* brow0 = wq + ((long)(cb + ml) * 9 + tap) * Cin + hi * 16;
    const _Float16* brow1 = brow0 + (long)16 * 9 * Cin;
    for (int cc = 0; cc < Cin; cc += 32) {
      V16U a, b0, b1;
      a.h[0] = *(const v8h*)(arow + cc);        // lanes<16: K0-7 | lanes>=16: K16-23
      a.h[1] = *(const v8h*)(arow + cc + 8);    // lanes<16: K8-15 | K24-31
      b0.v   = *(const v16h*)(brow0 + cc);      // lanes<16: K0-15 | K16-31
      b1.v   = *(const v16h*)(brow1 + cc);
      acc0 = __builtin_amdgcn_wmma_f32_16x16x32_f16(false, a.v, false, b0.v,
                                                    (short)0, acc0, false, false);
      acc1 = __builtin_amdgcn_wmma_f32_16x16x32_f16(false, a.v, false, b1.v,
                                                    (short)0, acc1, false, false);
    }
  }
  const float bs0 = bias[cb + ml];
  const float bs1 = bias[cb + 16 + ml];
#pragma unroll
  for (int r = 0; r < 8; ++r) {
    // D layout: VGPR r -> M = r (lanes 0-15) / r+8 (lanes 16-31), N = lane&15
    long pixoff = ((long)(h + 1) * WPD + (wx + r + hi * 8 + 1)) * ocs + ocoff + ml;
    float v0 = acc0[r] + bs0;
    float v1 = acc1[r] + bs1;
    if (relu) { v0 = v0 > 0.f ? v0 : 0.f; v1 = v1 > 0.f ? v1 : 0.f; }
    outp[pixoff + cb]      = (_Float16)v0;
    outp[pixoff + cb + 16] = (_Float16)v1;
  }
}

// ---------------- host orchestration ----------------
extern "C" void kernel_launch(void* const* d_in, const int* in_sizes, int n_in,
                              void* d_out, int out_size, void* d_ws, size_t ws_size,
                              hipStream_t stream) {
  const float* x  = (const float*)d_in[0];
  const float* w1 = (const float*)d_in[1];
  const float* b1 = (const float*)d_in[2];
  const float* wm = (const float*)d_in[3];
  const float* bm = (const float*)d_in[4];
  const float* wo = (const float*)d_in[5];
  const float* bo = (const float*)d_in[6];
  float* outF = (float*)d_out;
  char* ws = (char*)d_ws;

  size_t off = 0;
  auto alloc = [&](size_t bytes) { size_t o = off; off = (off + bytes + 511) & ~(size_t)511; return o; };

  const size_t szA   = (size_t)NPIX * 192 * 2;  // conv1 input (frames|state|down|pad)
  const size_t szB   = (size_t)NPIX * 128 * 2;  // ping
  const size_t szC   = (size_t)NPIX * 192 * 2;  // pong / out-conv result (192 ch stride)
  const size_t szOut = (size_t)3 * FH * FW * 4; // recurrent HR output (f32)
  const size_t szTmp = szOut;                   // blur temp
  const size_t szW1  = (size_t)128 * 9 * 192 * 2;
  const size_t szWm  = (size_t)5 * 128 * 9 * 128 * 2;
  const size_t szWo  = (size_t)192 * 9 * 128 * 2;
  const size_t szBia = 960 * 4;

  size_t oA = alloc(szA), oB = alloc(szB), oC = alloc(szC);
  size_t oOut = alloc(szOut);
  size_t zeroBytes = off;                       // zero A,B,C,out_buf every launch
  size_t oTmp = alloc(szTmp);
  size_t oW1 = alloc(szW1), oWm = alloc(szWm), oWo = alloc(szWo), oBias = alloc(szBia);
  (void)ws_size; (void)in_sizes; (void)n_in; (void)out_size;

  _Float16* actA = (_Float16*)(ws + oA);
  _Float16* actB = (_Float16*)(ws + oB);
  _Float16* actC = (_Float16*)(ws + oC);
  float* outBuf  = (float*)(ws + oOut);
  float* tmpBuf  = (float*)(ws + oTmp);
  _Float16* W1p  = (_Float16*)(ws + oW1);
  _Float16* Wmp  = (_Float16*)(ws + oWm);
  _Float16* Wop  = (_Float16*)(ws + oWo);
  float* Bias    = (float*)(ws + oBias);
  const float* Bc1 = Bias;
  const float* Bmp = Bias + 128;
  const float* Bop = Bias + 768;
  const long midWL = (long)128 * 9 * 128;

  // 1) zero activations + recurrent output (out0 = 0, state0 = 0, halos = 0)
  k_zero<<<4096, 256, 0, stream>>>((float4*)ws, (long)(zeroBytes / 16));

  // 2) pack weights/biases to WMMA-friendly f16 layouts
  k_pack_w1<<<512, 256, 0, stream>>>(w1, W1p);
  k_pack_wm<<<1024, 256, 0, stream>>>(wm, Wmp);
  k_pack_wo<<<512, 256, 0, stream>>>(wo, Wop);
  k_pack_bias<<<4, 256, 0, stream>>>(b1, bm, bo, Bias);

  dim3 blk(256);
  auto conv = [&](const _Float16* in, int ics, int Cin, const _Float16* wq, const float* bias,
                  _Float16* outp, int ocs, int ocoff, int coutPad, int relu) {
    dim3 grid(WID / 64, HGT, coutPad / 64);
    k_conv3x3_wmma<<<grid, blk, 0, stream>>>(in, ics, Cin, wq, bias, outp, ocs, ocoff, relu);
  };

  const int nHR = 3 * FH * FW;
  for (int t = 0; t < T_FRAMES; ++t) {
    // frames -> actA[0..8]
    k_pack_frames<<<(9 * HGT * WID + 255) / 256, blk, 0, stream>>>(x, actA, t);
    // shuffle_down(prev out): blur + clip + rint + reshape -> actA[137..184]
    k_blur_h<<<(nHR + 255) / 256, blk, 0, stream>>>(outBuf, tmpBuf);
    k_blur_vdown<<<(nHR + 255) / 256, blk, 0, stream>>>(tmpBuf, actA);
    // conv1 (185->128, cin padded to 192) + ReLU
    conv(actA, 192, 192, W1p, Bc1, actB, 128, 0, 128, 1);
    // 4 mid convs + ReLU (ping-pong B<->C; C uses stride 192, reads Cin=128)
    conv(actB, 128, 128, Wmp + 0 * midWL, Bmp + 0,   actC, 192, 0, 128, 1);
    conv(actC, 192, 128, Wmp + 1 * midWL, Bmp + 128, actB, 128, 0, 128, 1);
    conv(actB, 128, 128, Wmp + 2 * midWL, Bmp + 256, actC, 192, 0, 128, 1);
    conv(actC, 192, 128, Wmp + 3 * midWL, Bmp + 384, actB, 128, 0, 128, 1);
    // state conv (no ReLU) -> directly into next-step conv1 input (channels 9..136)
    conv(actB, 128, 128, Wmp + 4 * midWL, Bmp + 512, actA, 192, 9, 128, 0);
    // out conv 128 -> 176 (cout padded to 192), no ReLU -> actC
    conv(actB, 128, 128, Wop, Bop, actC, 192, 0, 192, 0);
    // res_add + shuffle_up -> recurrent HR buffer (final step -> d_out)
    float* dst = (t == T_FRAMES - 1) ? outF : outBuf;
    k_out_process<<<(HGT * WID + 255) / 256, blk, 0, stream>>>(actC, dst);
  }
}
